// TbNet_44573170598202
// MI455X (gfx1250) — matrix operand order repaired and checked by reading, not compiled, not run on hardware
//
#include <hip/hip_runtime.h>
#include <hip/hip_bf16.h>
#include <math.h>
#include <stdint.h>

#define N_NODES 100000
#define N_EDGES 1000000
#define F_IN 8
#define HID 64
#define VOCAB 64
#define TXT_F 64
#define SEQ 10
#define N_CLASSES 2

typedef __attribute__((ext_vector_type(16))) _Float16 v16h;
typedef __attribute__((ext_vector_type(8)))  _Float16 v8h;
typedef __attribute__((ext_vector_type(8)))  float    v8f;
typedef __attribute__((ext_vector_type(4)))  float    v4f;
typedef __attribute__((ext_vector_type(4)))  int      v4i;
typedef __attribute__((address_space(3))) v4i* lds_v4i_t;

// ---------------------------------------------------------------------------
// CDNA5 async global->LDS copy (ASYNCcnt) with safe fallback.
// Builtin signature (from hipcc diagnostic): param0 is `int __vector(4)*`.
// LDS destination produced via integer cast: on AMDGPU the low 32 bits of a
// generic LDS pointer are the LDS byte offset (flat->LDS truncates addr[31:0]).
// ---------------------------------------------------------------------------
__device__ __forceinline__ void async_copy_b128(_Float16* lds_dst,
                                                const _Float16* gsrc) {
#if __has_builtin(__builtin_amdgcn_global_load_async_to_lds_b128)
  __builtin_amdgcn_global_load_async_to_lds_b128(
      (v4i*)gsrc, (lds_v4i_t)(uint32_t)(uintptr_t)lds_dst, 0, 0);
#else
  *(v8h*)lds_dst = *(const v8h*)gsrc;
#endif
}

__device__ __forceinline__ void wait_async() {
#if __has_builtin(__builtin_amdgcn_s_wait_asynccnt)
  __builtin_amdgcn_s_wait_asynccnt(0);
#else
  asm volatile("s_wait_asynccnt 0" ::: "memory");
#endif
}

// ---------------------------------------------------------------------------
// WMMA helpers (CDNA5 wave32, V_WMMA_F32_16X16X32_F16)
// A fragment: row-major f16 tile [16 x ldk] in LDS.
// B fragment: torch-layout weights W[O x K] row-major f16; computes A @ W^T.
// ---------------------------------------------------------------------------
__device__ __forceinline__ v16h pack16(v8h lo, v8h hi) {
  v16h r;
#pragma unroll
  for (int i = 0; i < 8; ++i) { r[i] = lo[i]; r[i + 8] = hi[i]; }
  return r;
}

__device__ __forceinline__ v16h a_frag(const _Float16* A, int ldk, int kb) {
  int lane = threadIdx.x & 31;
  int row = lane & 15, hi = lane >> 4;
  const _Float16* p = A + row * ldk + kb;
  v8h lo = *(const v8h*)(p + 8 * hi);
  v8h hh = *(const v8h*)(p + 16 + 8 * hi);
  return pack16(lo, hh);
}

__device__ __forceinline__ v16h b_frag(const _Float16* W, int ldk, int o0, int kb) {
  int lane = threadIdx.x & 31;
  int col = lane & 15, hi = lane >> 4;
  const _Float16* p = W + (o0 + col) * ldk + kb + 8 * hi;
  v8h lo = *(const v8h*)p;
  v8h hh = *(const v8h*)(p + 16);
  return pack16(lo, hh);
}

__device__ __forceinline__ v8f wmma_f16(v16h a, v16h b, v8f c) {
  return __builtin_amdgcn_wmma_f32_16x16x32_f16(false, a, false, b, (short)0, c,
                                                false, false);
}

__device__ __forceinline__ float fast_sigmoid(float x) {
  return __builtin_amdgcn_rcpf(1.f + __expf(-x));
}
__device__ __forceinline__ float fast_tanh(float x) {
  float e2 = __expf(2.f * x);
  return (e2 - 1.f) * __builtin_amdgcn_rcpf(e2 + 1.f);
}

// ---------------------------------------------------------------------------
// Small utility kernels
// ---------------------------------------------------------------------------
__global__ void k_fill(float* p, int n, float v) {
  int i = blockIdx.x * blockDim.x + threadIdx.x;
  if (i < n) p[i] = v;
}

__global__ void k_deg(const int* __restrict__ dst, float* __restrict__ deg) {
  int e = blockIdx.x * blockDim.x + threadIdx.x;
  if (e < N_EDGES) atomicAdd(&deg[dst[e]], 1.0f);
}

__global__ void k_dinv(const float* __restrict__ deg, float* __restrict__ dinv) {
  int i = blockIdx.x * blockDim.x + threadIdx.x;
  if (i < N_NODES) {
    float d = deg[i];
    dinv[i] = d > 0.f ? rsqrtf(d) : 0.f;
  }
}

__global__ void k_norm(const int* __restrict__ src, const int* __restrict__ dst,
                       const float* __restrict__ dinv, float* __restrict__ norm) {
  int e = blockIdx.x * blockDim.x + threadIdx.x;
  if (e < N_EDGES) norm[e] = dinv[src[e]] * dinv[dst[e]];
}

// GCN layer 1 linear: [N,8] @ [8,64]; K=8, scalar VALU is fine (51 MFLOP).
__global__ void k_gcn1_gemm(const float* __restrict__ x, const float* __restrict__ W,
                            float* __restrict__ out) {
  __shared__ float sW[F_IN * HID];
  for (int i = threadIdx.x; i < F_IN * HID; i += blockDim.x) sW[i] = W[i];
  __syncthreads();
  int i = blockIdx.x * blockDim.x + threadIdx.x;
  if (i >= N_NODES * HID) return;
  int n = i >> 6, j = i & 63;
  float s = 0.f;
#pragma unroll
  for (int f = 0; f < F_IN; ++f) s += x[n * F_IN + f] * sW[f * HID + j];
  out[i] = s;
}

// Edge scatter: agg[dst] += hlin[src] * norm  (grid-stride over E*64)
__global__ void k_scatter(const int* __restrict__ src, const int* __restrict__ dst,
                          const float* __restrict__ norm,
                          const float* __restrict__ hlin, float* __restrict__ agg) {
  const long long total = (long long)N_EDGES * HID;
  long long stride = (long long)gridDim.x * blockDim.x;
  for (long long i = (long long)blockIdx.x * blockDim.x + threadIdx.x; i < total;
       i += stride) {
    int e = (int)(i >> 6), j = (int)(i & 63);
    atomicAdd(&agg[dst[e] * HID + j], hlin[src[e] * HID + j] * norm[e]);
  }
}

// h = relu(agg + self_loop + bias), stored f16 (consumed only as WMMA A-operand)
__global__ void k_finalize(const float* __restrict__ agg, const float* __restrict__ hlin,
                           const float* __restrict__ dinv, const float* __restrict__ b,
                           _Float16* __restrict__ h) {
  int i = blockIdx.x * blockDim.x + threadIdx.x;
  if (i >= N_NODES * HID) return;
  int n = i >> 6, j = i & 63;
  float dv = dinv[n];
  float v = agg[i] + hlin[i] * dv * dv + b[j];
  h[i] = (_Float16)fmaxf(v, 0.f);
}

// ---------------------------------------------------------------------------
// GCN layer 2 linear via WMMA: [N,64](f16) @ [64,64] -> f32
// block = 128 (4 waves), wave = one 16-node tile. A staged via async-to-LDS,
// output staged through LDS for coalesced b128 stores.
// ---------------------------------------------------------------------------
__global__ __launch_bounds__(128) void k_gcn2_gemm(const _Float16* __restrict__ h1,
                                                   const float* __restrict__ W,
                                                   float* __restrict__ out) {
  __shared__ __align__(16) _Float16 sW[HID * HID];      // sW[o][k] = W[k][o]
  __shared__ __align__(16) _Float16 sA[4][16 * HID];
  __shared__ __align__(16) float sOut[4][16 * HID];
  int tid = threadIdx.x;
  for (int i = tid; i < HID * HID; i += blockDim.x) {
    int o = i >> 6, k = i & 63;
    sW[i] = (_Float16)W[k * HID + o];
  }
  int wave = tid >> 5, lane = tid & 31;
  int colin = lane & 15, hi = lane >> 4;
  int rowbase = (blockIdx.x * 4 + wave) * 16;
  _Float16* A = sA[wave];
  // async gather: 16 rows x 64 halves = 128 chunks of 16B
#pragma unroll
  for (int it = 0; it < 4; ++it) {
    int chunk = lane + 32 * it;
    int r = chunk >> 3, seg = (chunk & 7) * 8;
    int n = rowbase + r;
    if (n >= N_NODES) n = N_NODES - 1;
    async_copy_b128(A + r * HID + seg, h1 + n * HID + seg);
  }
  wait_async();
  __syncthreads();

  v8f acc[4];
#pragma unroll
  for (int ct = 0; ct < 4; ++ct)
#pragma unroll
    for (int i = 0; i < 8; ++i) acc[ct][i] = 0.f;
#pragma unroll
  for (int kt = 0; kt < 2; ++kt) {
    v16h a = a_frag(A, HID, kt * 32);
#pragma unroll
    for (int ct = 0; ct < 4; ++ct)
      acc[ct] = wmma_f16(a, b_frag(sW, HID, ct * 16, kt * 32), acc[ct]);
  }
  // D tiles -> LDS (row-major f32), then coalesced b128 row stores
  float* O = sOut[wave];
#pragma unroll
  for (int ct = 0; ct < 4; ++ct)
#pragma unroll
    for (int r = 0; r < 8; ++r)
      O[(r + hi * 8) * HID + ct * 16 + colin] = acc[ct][r];
  __syncthreads();
#pragma unroll
  for (int it = 0; it < 8; ++it) {
    int chunk = lane + 32 * it;          // 256 chunks of 4 floats
    int r = chunk >> 4, seg = (chunk & 15) * 4;
    int n = rowbase + r;
    if (n < N_NODES) *(v4f*)(out + n * HID + seg) = *(const v4f*)(O + r * HID + seg);
  }
}

// ---------------------------------------------------------------------------
// Fused embedding + GRU (T=10, H=64). Weights + embedding in LDS (f16).
// Hidden state lives in WMMA C/D registers; round-trips through a per-wave
// LDS f16 tile to become next step's A operand. Output written f16.
// ---------------------------------------------------------------------------
__global__ __launch_bounds__(128) void k_gru(
    const int* __restrict__ xtext, const float* __restrict__ embed,
    const float* __restrict__ w_ih, const float* __restrict__ w_hh,
    const float* __restrict__ b_ih, const float* __restrict__ b_hh,
    _Float16* __restrict__ textout) {
  __shared__ __align__(16) _Float16 sWih[3 * HID * TXT_F];  // [192][64]
  __shared__ __align__(16) _Float16 sWhh[3 * HID * HID];    // [192][64]
  __shared__ __align__(16) _Float16 sEmb[VOCAB * TXT_F];
  __shared__ __align__(16) _Float16 sXe[4][16 * TXT_F];
  __shared__ __align__(16) _Float16 sH[4][16 * HID];
  int tid = threadIdx.x;
  for (int i = tid; i < 3 * HID * TXT_F; i += blockDim.x) sWih[i] = (_Float16)w_ih[i];
  for (int i = tid; i < 3 * HID * HID; i += blockDim.x) sWhh[i] = (_Float16)w_hh[i];
  for (int i = tid; i < VOCAB * TXT_F; i += blockDim.x) sEmb[i] = (_Float16)embed[i];
  int wave = tid >> 5, lane = tid & 31;
  int colin = lane & 15, hi = lane >> 4;
  _Float16* Xe = sXe[wave];
  _Float16* Hb = sH[wave];
  for (int i = lane; i < 16 * HID; i += 32) Hb[i] = (_Float16)0.f;
  int rowbase = (blockIdx.x * 4 + wave) * 16;
  __syncthreads();

  v8f h[4];
#pragma unroll
  for (int c = 0; c < 4; ++c)
#pragma unroll
    for (int i = 0; i < 8; ++i) h[c][i] = 0.f;

  for (int t = 0; t < SEQ; ++t) {
    // stage this step's embedded tokens: Xe[16 x 64] f16
    for (int i = lane; i < 16 * TXT_F; i += 32) {
      int r = i >> 6, f = i & 63;
      int n = rowbase + r;
      if (n >= N_NODES) n = N_NODES - 1;
      int tok = xtext[n * SEQ + t];
      Xe[i] = sEmb[tok * TXT_F + f];
    }
    __syncthreads();

    v8f gi[12], gh[12];
#pragma unroll
    for (int ct = 0; ct < 12; ++ct) {
      float bi = b_ih[ct * 16 + colin];
      float bh = b_hh[ct * 16 + colin];
#pragma unroll
      for (int i = 0; i < 8; ++i) { gi[ct][i] = bi; gh[ct][i] = bh; }
    }
#pragma unroll
    for (int kt = 0; kt < 2; ++kt) {
      v16h ax = a_frag(Xe, TXT_F, kt * 32);
      v16h ah = a_frag(Hb, HID, kt * 32);
#pragma unroll
      for (int ct = 0; ct < 12; ++ct) {
        gi[ct] = wmma_f16(ax, b_frag(sWih, TXT_F, ct * 16, kt * 32), gi[ct]);
        gh[ct] = wmma_f16(ah, b_frag(sWhh, HID, ct * 16, kt * 32), gh[ct]);
      }
    }
    __syncthreads();  // done reading Hb, safe to overwrite
    // gate math directly in C/D layout (tiles c / c+4 / c+8 share (lane,vgpr))
#pragma unroll
    for (int c = 0; c < 4; ++c)
#pragma unroll
      for (int i = 0; i < 8; ++i) {
        float r = fast_sigmoid(gi[c][i] + gh[c][i]);
        float z = fast_sigmoid(gi[c + 4][i] + gh[c + 4][i]);
        float nn = fast_tanh(gi[c + 8][i] + r * gh[c + 8][i]);
        h[c][i] = (1.f - z) * nn + z * h[c][i];
      }
    // h -> per-wave LDS tile (f16 row-major) for next step's A operand
#pragma unroll
    for (int c = 0; c < 4; ++c)
#pragma unroll
      for (int r = 0; r < 8; ++r)
        Hb[(r + hi * 8) * HID + c * 16 + colin] = (_Float16)h[c][r];
    __syncthreads();
  }
  // Hb already holds final h row-major f16 -> coalesced b128 stores
#pragma unroll
  for (int it = 0; it < 4; ++it) {
    int chunk = lane + 32 * it;          // 128 chunks of 8 halves
    int r = chunk >> 3, seg = (chunk & 7) * 8;
    int n = rowbase + r;
    if (n < N_NODES) *(v8h*)(textout + n * HID + seg) = *(const v8h*)(Hb + r * HID + seg);
  }
}

// ---------------------------------------------------------------------------
// Fused pair MLPs + final head + log_softmax over edges.
// Gathers concat(h[s],h[t]) f16 rows via async-to-LDS, WMMA GEMMs against
// LDS f16 weights, scalar 128-wide head with lane-16 shuffle for the lse.
// ---------------------------------------------------------------------------
__global__ __launch_bounds__(128) void k_pair(
    const int* __restrict__ ei, const _Float16* __restrict__ h2,
    const _Float16* __restrict__ textout, const float* __restrict__ lin1_w,
    const float* __restrict__ lin1_b, const float* __restrict__ lint_w,
    const float* __restrict__ lint_b, const float* __restrict__ linf_w,
    const float* __restrict__ linf_b, float* __restrict__ out) {
  __shared__ __align__(16) _Float16 sW1[HID * 2 * HID];  // [64][128] torch layout
  __shared__ __align__(16) _Float16 sWt[HID * 2 * HID];
  __shared__ float sWf[N_CLASSES * 2 * HID];             // [2][128]
  __shared__ __align__(16) _Float16 sA[4][16 * 2 * HID]; // per-wave gather tile
  __shared__ __align__(16) _Float16 sP[4][16 * 2 * HID]; // xpair|xpairtext f16
  int tid = threadIdx.x;
  for (int i = tid; i < HID * 2 * HID; i += blockDim.x) {
    sW1[i] = (_Float16)lin1_w[i];
    sWt[i] = (_Float16)lint_w[i];
  }
  for (int i = tid; i < N_CLASSES * 2 * HID; i += blockDim.x) sWf[i] = linf_w[i];

  int wave = tid >> 5, lane = tid & 31;
  int colin = lane & 15, hi = lane >> 4;
  int ebase = (blockIdx.x * 4 + wave) * 16;
  const int* src = ei;
  const int* dst = ei + N_EDGES;
  _Float16* A = sA[wave];
  _Float16* P = sP[wave];

  // ---- stage 1: xpair = relu([h2[s]|h2[t]] @ lin1_w^T + b1) ----
  // async gather: 16 rows x 128 halves = 256 chunks of 16B
#pragma unroll
  for (int it = 0; it < 8; ++it) {
    int chunk = lane + 32 * it;
    int r = chunk >> 4, kk = (chunk & 15) * 8;   // half-offset 0..120
    int e = ebase + r;
    if (e >= N_EDGES) e = N_EDGES - 1;
    int n = (kk < HID) ? src[e] : dst[e];
    async_copy_b128(A + r * 128 + kk, h2 + n * HID + (kk & 63));
  }
  wait_async();
  __syncthreads();   // also covers the weight staging above

  v8f acc[4];
#pragma unroll
  for (int ct = 0; ct < 4; ++ct) {
    float bv = lin1_b[ct * 16 + colin];
#pragma unroll
    for (int i = 0; i < 8; ++i) acc[ct][i] = bv;
  }
#pragma unroll
  for (int kt = 0; kt < 4; ++kt) {
    v16h a = a_frag(A, 128, kt * 32);
#pragma unroll
    for (int ct = 0; ct < 4; ++ct)
      acc[ct] = wmma_f16(a, b_frag(sW1, 128, ct * 16, kt * 32), acc[ct]);
  }
#pragma unroll
  for (int ct = 0; ct < 4; ++ct)
#pragma unroll
    for (int r = 0; r < 8; ++r)
      P[(r + hi * 8) * 128 + ct * 16 + colin] = (_Float16)fmaxf(acc[ct][r], 0.f);
  __syncthreads();  // A may be overwritten now

  // ---- stage 2: xpairtext = relu([text[s]|text[t]] @ lin_text_w^T + bt) ----
#pragma unroll
  for (int it = 0; it < 8; ++it) {
    int chunk = lane + 32 * it;
    int r = chunk >> 4, kk = (chunk & 15) * 8;
    int e = ebase + r;
    if (e >= N_EDGES) e = N_EDGES - 1;
    int n = (kk < HID) ? src[e] : dst[e];
    async_copy_b128(A + r * 128 + kk, textout + n * HID + (kk & 63));
  }
  wait_async();
  __syncthreads();
#pragma unroll
  for (int ct = 0; ct < 4; ++ct) {
    float bv = lint_b[ct * 16 + colin];
#pragma unroll
    for (int i = 0; i < 8; ++i) acc[ct][i] = bv;
  }
#pragma unroll
  for (int kt = 0; kt < 4; ++kt) {
    v16h a = a_frag(A, 128, kt * 32);
#pragma unroll
    for (int ct = 0; ct < 4; ++ct)
      acc[ct] = wmma_f16(a, b_frag(sWt, 128, ct * 16, kt * 32), acc[ct]);
  }
#pragma unroll
  for (int ct = 0; ct < 4; ++ct)
#pragma unroll
    for (int r = 0; r < 8; ++r)
      P[(r + hi * 8) * 128 + 64 + ct * 16 + colin] = (_Float16)fmaxf(acc[ct][r], 0.f);
  __syncthreads();

  // ---- final head: [16 x 128] . [2 x 128]^T + log_softmax over 2 classes ----
  int row = lane & 15;
  int cls = lane >> 4;
  const float* wf0 = sWf + cls * 128;               // xpair part of concat
  const float* wf1 = wf0 + 64;                      // xpairtext part
  float s = linf_b[cls];
  const _Float16* Pr = P + row * 128;
#pragma unroll 8
  for (int k = 0; k < 64; ++k) s += (float)Pr[k] * wf0[k];
#pragma unroll 8
  for (int k = 0; k < 64; ++k) s += (float)Pr[64 + k] * wf1[k];
  float other = __shfl_xor(s, 16, 32);
  float m = fmaxf(s, other);
  float lse = m + __logf(__expf(s - m) + __expf(other - m));
  int e = ebase + row;
  if (e < N_EDGES) out[e * N_CLASSES + cls] = s - lse;
}

// ---------------------------------------------------------------------------
// Launch
// ---------------------------------------------------------------------------
extern "C" void kernel_launch(void* const* d_in, const int* in_sizes, int n_in,
                              void* d_out, int out_size, void* d_ws, size_t ws_size,
                              hipStream_t stream) {
  const float* x       = (const float*)d_in[0];
  const int*   ei      = (const int*)d_in[1];
  const int*   xtext   = (const int*)d_in[2];
  const float* conv1_w = (const float*)d_in[3];
  const float* conv1_b = (const float*)d_in[4];
  const float* conv2_w = (const float*)d_in[5];
  const float* conv2_b = (const float*)d_in[6];
  const float* embed   = (const float*)d_in[7];
  const float* w_ih    = (const float*)d_in[8];
  const float* w_hh    = (const float*)d_in[9];
  const float* b_ih    = (const float*)d_in[10];
  const float* b_hh    = (const float*)d_in[11];
  const float* lin1_w  = (const float*)d_in[12];
  const float* lin1_b  = (const float*)d_in[13];
  const float* lint_w  = (const float*)d_in[14];
  const float* lint_b  = (const float*)d_in[15];
  const float* linf_w  = (const float*)d_in[16];
  const float* linf_b  = (const float*)d_in[17];
  float* out = (float*)d_out;

  const int* src = ei;
  const int* dst = ei + N_EDGES;

  // workspace layout:
  //   f32: deg[N] | dinv[N] | norm[E] | lin[N*64] | agg[N*64]
  //   f16: h1h[N*64] | h2h[N*64] | texth[N*64]
  float* ws   = (float*)d_ws;
  float* deg  = ws;
  float* dinv = deg + N_NODES;
  float* norm = dinv + N_NODES;
  float* lin  = norm + N_EDGES;        // hlin1 then hlin2 (f32)
  float* agg  = lin + N_NODES * HID;
  _Float16* h1h   = (_Float16*)(agg + N_NODES * HID);
  _Float16* h2h   = h1h + N_NODES * HID;
  _Float16* texth = h2h + N_NODES * HID;
  (void)ws_size; (void)in_sizes; (void)n_in; (void)out_size;

  const int T = 256;
  const int gN   = (N_NODES + T - 1) / T;
  const int gE   = (N_EDGES + T - 1) / T;
  const int gNH  = (N_NODES * HID + T - 1) / T;
  const int gT16 = (N_NODES + 63) / 64;   // 4 waves * 16 rows per block
  const int gE16 = (N_EDGES + 63) / 64;

  // GCN normalization
  k_fill<<<gN, T, 0, stream>>>(deg, N_NODES, 1.0f);         // self loops
  k_deg<<<gE, T, 0, stream>>>(dst, deg);
  k_dinv<<<gN, T, 0, stream>>>(deg, dinv);
  k_norm<<<gE, T, 0, stream>>>(src, dst, dinv, norm);

  // GCN layer 1
  k_gcn1_gemm<<<gNH, T, 0, stream>>>(x, conv1_w, lin);
  k_fill<<<gNH, T, 0, stream>>>(agg, N_NODES * HID, 0.0f);
  k_scatter<<<8192, T, 0, stream>>>(src, dst, norm, lin, agg);
  k_finalize<<<gNH, T, 0, stream>>>(agg, lin, dinv, conv1_b, h1h);   // h1 (f16)

  // GCN layer 2 (WMMA linear on f16 h1)
  k_gcn2_gemm<<<gT16, 128, 0, stream>>>(h1h, conv2_w, lin);          // hlin2 f32
  k_fill<<<gNH, T, 0, stream>>>(agg, N_NODES * HID, 0.0f);
  k_scatter<<<8192, T, 0, stream>>>(src, dst, norm, lin, agg);
  k_finalize<<<gNH, T, 0, stream>>>(agg, lin, dinv, conv2_b, h2h);   // h2 (f16)

  // Embedding + GRU (WMMA, fused), f16 output
  k_gru<<<gT16, 128, 0, stream>>>(xtext, embed, w_ih, w_hh, b_ih, b_hh, texth);

  // Pair MLPs + head + log_softmax (WMMA + async gathers)
  k_pair<<<gE16, 128, 0, stream>>>(ei, h2h, texth, lin1_w, lin1_b, lint_w,
                                   lint_b, linf_w, linf_b, out);
}
